// MultiHopRelationAttention_40879498729276
// MI455X (gfx1250) — matrix operand order repaired and checked by reading, not compiled, not run on hardware
//
#include <hip/hip_runtime.h>

// ---------------------------------------------------------------------------
// MultiHopRelationAttention for MI455X (gfx1250, wave32, WMMA bf16 16x16x32)
// B=4, T=2048, D=768, H=12, HD=64, R=64
// ---------------------------------------------------------------------------

#define Bc  4
#define Tc  2048
#define Dc  768
#define Hc  12
#define HDc 64
#define BTc (Bc * Tc)      // 8192 tokens
#define Rc  64

typedef __bf16 v8bf  __attribute__((ext_vector_type(8)));
typedef __bf16 v16bf __attribute__((ext_vector_type(16)));
typedef float  v8f   __attribute__((ext_vector_type(8)));

static __device__ __forceinline__ v8bf ld8(const __bf16* p) {
    return *(const v8bf*)p;
}
static __device__ __forceinline__ v16bf cat8(v8bf lo, v8bf hi) {
    v16bf r;
#pragma unroll
    for (int i = 0; i < 8; ++i) { r[i] = lo[i]; r[8 + i] = hi[i]; }
    return r;
}
static __device__ __forceinline__ v8f wmma_bf16(v16bf a, v16bf b, v8f c) {
    return __builtin_amdgcn_wmma_f32_16x16x32_bf16(
        /*neg_a=*/false, a, /*neg_b=*/false, b,
        /*c_mod=*/(short)0, c, /*reuse_a=*/false, /*reuse_b=*/false);
}
// Reductions across the 16 lanes of one wave32 half (xor masks 1,2,4,8 keep
// lane/16 invariant -> rows of the C-layout tile reduce independently).
static __device__ __forceinline__ float redmax16(float v) {
#pragma unroll
    for (int m = 1; m < 16; m <<= 1) v = fmaxf(v, __shfl_xor(v, m, 32));
    return v;
}
static __device__ __forceinline__ float redsum16(float v) {
#pragma unroll
    for (int m = 1; m < 16; m <<= 1) v += __shfl_xor(v, m, 32);
    return v;
}

// ---------------------------------------------------------------------------
// softmax over 3 path weights
// ---------------------------------------------------------------------------
__global__ void mhra_softmax3_kernel(const float* __restrict__ pw,
                                     float* __restrict__ wsm) {
    if (threadIdx.x == 0 && blockIdx.x == 0) {
        float a = pw[0], b = pw[1], c = pw[2];
        float m = fmaxf(a, fmaxf(b, c));
        float e0 = __expf(a - m), e1 = __expf(b - m), e2 = __expf(c - m);
        float s = e0 + e1 + e2;
        wsm[0] = e0 / s; wsm[1] = e1 / s; wsm[2] = e2 / s;
    }
}

// ---------------------------------------------------------------------------
// Per-batch relation histogram -> collapsed row bias biasA[b,i]
// ---------------------------------------------------------------------------
__global__ void mhra_bias_prep_kernel(const int* __restrict__ rel,
                                      const float* __restrict__ hop0,
                                      const float* __restrict__ hop1,
                                      const float* __restrict__ wsm,
                                      float* __restrict__ biasA) {
    __shared__ int hist[Rc];
    const int bb = blockIdx.x;
    for (int i = threadIdx.x; i < Rc; i += blockDim.x) hist[i] = 0;
    __syncthreads();
    for (int j = threadIdx.x; j < Tc; j += blockDim.x)
        atomicAdd(&hist[rel[bb * Tc + j] & (Rc - 1)], 1);
    __syncthreads();
    const float w0 = wsm[0], w1 = wsm[1];
    for (int i = threadIdx.x; i < Tc; i += blockDim.x) {
        int r = rel[bb * Tc + i] & (Rc - 1);
        float ba = 0.f;
        if (r > 0) {
            float b0 = hop0[r], b1 = hop1[r];
            ba = w0 * b0;
            if (hist[r] >= 3) ba += 0.5f * w1 * (b0 + b1);
        }
        biasA[bb * Tc + i] = ba;
    }
}

// ---------------------------------------------------------------------------
// f32 -> bf16 elementwise convert
// ---------------------------------------------------------------------------
__global__ void mhra_cvt_bf16_kernel(const float* __restrict__ in,
                                     __bf16* __restrict__ out, int n) {
    int i = blockIdx.x * blockDim.x + threadIdx.x;
    if (i < n) out[i] = (__bf16)in[i];
}

// ---------------------------------------------------------------------------
// GEMM  Y = X(bf16 [N,768]) @ W(bf16 [M,768])^T + bias(f32 [M])
// block = 128 threads = 4 waves; block tile 128(rows) x 64(cols);
// wave w -> 32 rows (2 A tiles) x 64 cols (4 B tiles) = 8 WMMA accumulators.
// K = 768 is compile-time: fully unrolled, software-pipelined (prefetch k+32
// while current WMMAs run), last chunk peeled so the steady state is
// branch-free with immediate-offset b128 loads.
// mode 0: write bf16 into [B,H,T,HD] (QKV)   mode 1: write f32 flat [N,M]
// ---------------------------------------------------------------------------
__global__ void __launch_bounds__(128)
mhra_gemm_kernel(const __bf16* __restrict__ X, const __bf16* __restrict__ W,
                 const float* __restrict__ bias, __bf16* __restrict__ outBf,
                 float* __restrict__ outF, int mode) {
    constexpr int KD = Dc;  // 768
    const int wave = threadIdx.x >> 5, lane = threadIdx.x & 31;
    const int hf = lane >> 4, n16 = lane & 15;
    const int rowBase = blockIdx.x * 128 + wave * 32;
    const int mbase   = blockIdx.y * 64;

    v8f acc[2][4];
#pragma unroll
    for (int r = 0; r < 2; ++r)
#pragma unroll
        for (int t = 0; t < 4; ++t) acc[r][t] = 0.f;

    // Per-lane base pointers (k offsets become instruction immediates).
    const __bf16* pa0 = X + (size_t)(rowBase + n16) * KD + 16 * hf;
    const __bf16* pa1 = X + (size_t)(rowBase + 16 + n16) * KD + 16 * hf;
    const __bf16* pb0 = W + (size_t)(mbase + 0  + n16) * KD + 16 * hf;
    const __bf16* pb1 = W + (size_t)(mbase + 16 + n16) * KD + 16 * hf;
    const __bf16* pb2 = W + (size_t)(mbase + 32 + n16) * KD + 16 * hf;
    const __bf16* pb3 = W + (size_t)(mbase + 48 + n16) * KD + 16 * hf;

    // A 16x32 layout: per-lane 16 contiguous K at 16*hf
    auto ldA = [&](const __bf16* p, int k0) {
        return cat8(ld8(p + k0), ld8(p + k0 + 8));
    };
    // B 32x16 layout: lane = out col (W row), 16 contiguous K at 16*hf
    auto ldB = [&](const __bf16* p, int k0) {
        return cat8(ld8(p + k0), ld8(p + k0 + 8));
    };

    v16bf a0 = ldA(pa0, 0), a1 = ldA(pa1, 0);
    v16bf b0 = ldB(pb0, 0), b1 = ldB(pb1, 0);
    v16bf b2 = ldB(pb2, 0), b3 = ldB(pb3, 0);

#pragma unroll
    for (int k0 = 0; k0 < KD - 32; k0 += 32) {
        // prefetch next chunk into fresh registers (immediate offsets)
        v16bf a0n = ldA(pa0, k0 + 32), a1n = ldA(pa1, k0 + 32);
        v16bf b0n = ldB(pb0, k0 + 32), b1n = ldB(pb1, k0 + 32);
        v16bf b2n = ldB(pb2, k0 + 32), b3n = ldB(pb3, k0 + 32);

        acc[0][0] = wmma_bf16(a0, b0, acc[0][0]);
        acc[1][0] = wmma_bf16(a1, b0, acc[1][0]);
        acc[0][1] = wmma_bf16(a0, b1, acc[0][1]);
        acc[1][1] = wmma_bf16(a1, b1, acc[1][1]);
        acc[0][2] = wmma_bf16(a0, b2, acc[0][2]);
        acc[1][2] = wmma_bf16(a1, b2, acc[1][2]);
        acc[0][3] = wmma_bf16(a0, b3, acc[0][3]);
        acc[1][3] = wmma_bf16(a1, b3, acc[1][3]);

        a0 = a0n; a1 = a1n;
        b0 = b0n; b1 = b1n; b2 = b2n; b3 = b3n;
    }
    // peeled final chunk
    acc[0][0] = wmma_bf16(a0, b0, acc[0][0]);
    acc[1][0] = wmma_bf16(a1, b0, acc[1][0]);
    acc[0][1] = wmma_bf16(a0, b1, acc[0][1]);
    acc[1][1] = wmma_bf16(a1, b1, acc[1][1]);
    acc[0][2] = wmma_bf16(a0, b2, acc[0][2]);
    acc[1][2] = wmma_bf16(a1, b2, acc[1][2]);
    acc[0][3] = wmma_bf16(a0, b3, acc[0][3]);
    acc[1][3] = wmma_bf16(a1, b3, acc[1][3]);

#pragma unroll
    for (int r = 0; r < 2; ++r) {
#pragma unroll
        for (int t = 0; t < 4; ++t) {
#pragma unroll
            for (int e = 0; e < 8; ++e) {
                int m   = mbase + t * 16 + n16;                   // C col
                int tok = rowBase + r * 16 + e + 8 * hf;          // C row
                float val = acc[r][t][e] + bias[m];
                if (mode == 0) {
                    int bb = tok >> 11, tt = tok & (Tc - 1);
                    int hh = m >> 6, dd = m & 63;
                    outBf[(((size_t)bb * Hc + hh) * Tc + tt) * HDc + dd] =
                        (__bf16)val;
                } else {
                    outF[(size_t)tok * Dc + m] = val;
                }
            }
        }
    }
}

// ---------------------------------------------------------------------------
// Flash attention per (b,h): block = 128 thr = 4 waves; block owns 64 query
// rows (16 per wave); streams K/V in chunks of 32 keys. S = Q K^T via 2x2
// WMMAs, online softmax with per-half shuffle reductions, P -> LDS -> A
// layout, PV via 4 WMMAs against LDS-transposed V (double-buffered, one
// barrier per chunk).
// ---------------------------------------------------------------------------
__global__ void __launch_bounds__(128)
mhra_flash_attn_kernel(const __bf16* __restrict__ Q, const __bf16* __restrict__ Kx,
                       const __bf16* __restrict__ V, const int* __restrict__ rel,
                       const int* __restrict__ amask,
                       const float* __restrict__ biasA, __bf16* __restrict__ O) {
    __shared__ __bf16 Vt[2][HDc * 32];   // V transposed [d][j], 2 x 4 KB
    __shared__ __bf16 Pst[4 * 16 * 32];  // per-wave P staging, 4 KB

    const int wave = threadIdx.x >> 5, lane = threadIdx.x & 31;
    const int hf = lane >> 4, n16 = lane & 15;
    const int bh = blockIdx.y, bb = bh / Hc, hh = bh - bb * Hc;
    const int qbase = blockIdx.x * 64 + wave * 16;
    const float scale = 0.125f;  // 1/sqrt(64)

    // Q in A-layout, both K-chunks (HD=64 -> 2 x 16x32 A tiles)
    const __bf16* Qp = Q + ((size_t)bh * Tc + qbase + n16) * HDc;
    v16bf aq0 = cat8(ld8(Qp + 16 * hf), ld8(Qp + 16 * hf + 8));
    v16bf aq1 = cat8(ld8(Qp + 32 + 16 * hf), ld8(Qp + 40 + 16 * hf));

    int   ri[8];  float ba[8];  int itok[8];
    float mr[8], lr[8];
    v8f acc[4];
#pragma unroll
    for (int t = 0; t < 4; ++t) acc[t] = 0.f;
#pragma unroll
    for (int e = 0; e < 8; ++e) {
        itok[e] = qbase + 8 * hf + e;                 // C-layout row
        ri[e] = rel[bb * Tc + itok[e]];
        ba[e] = biasA[bb * Tc + itok[e]];
        mr[e] = -1e30f; lr[e] = 0.f;
    }

    int buf = 0;
    for (int j0 = 0; j0 < Tc; j0 += 32, buf ^= 1) {
        // cooperative transposed V tile load: Vt[buf][d][j]
        {
            int jl = threadIdx.x >> 2;   // 0..31
            int sg = threadIdx.x & 3;    // 0..3 (16-wide d segment)
            const __bf16* vp = V + ((size_t)bh * Tc + j0 + jl) * HDc + sg * 16;
            v8bf v0 = ld8(vp), v1 = ld8(vp + 8);
#pragma unroll
            for (int q = 0; q < 8; ++q) {
                Vt[buf][(sg * 16 + q) * 32 + jl]     = v0[q];
                Vt[buf][(sg * 16 + 8 + q) * 32 + jl] = v1[q];
            }
        }
        __syncthreads();  // also orders WAR on the alternate buffer

        int rj0 = rel[bb * Tc + j0 + n16],   rj1 = rel[bb * Tc + j0 + 16 + n16];
        int mk0 = amask[bb * Tc + j0 + n16], mk1 = amask[bb * Tc + j0 + 16 + n16];

        // S tiles: keys [j0..j0+16) and [j0+16..j0+32); B-layout straight from
        // global K (lane = key, 16 contiguous d at 32c + 16*hf)
        const __bf16* Kp0 = Kx + ((size_t)bh * Tc + j0 + n16) * HDc;
        const __bf16* Kp1 = Kx + ((size_t)bh * Tc + j0 + 16 + n16) * HDc;
        v8f s0 = 0.f, s1 = 0.f;
        s0 = wmma_bf16(aq0, cat8(ld8(Kp0 + 16 * hf), ld8(Kp0 + 16 * hf + 8)), s0);
        s0 = wmma_bf16(aq1, cat8(ld8(Kp0 + 32 + 16 * hf), ld8(Kp0 + 40 + 16 * hf)), s0);
        s1 = wmma_bf16(aq0, cat8(ld8(Kp1 + 16 * hf), ld8(Kp1 + 16 * hf + 8)), s1);
        s1 = wmma_bf16(aq1, cat8(ld8(Kp1 + 32 + 16 * hf), ld8(Kp1 + 40 + 16 * hf)), s1);

        // online softmax per row (row = e + 8*hf, reductions within half-wave)
#pragma unroll
        for (int e = 0; e < 8; ++e) {
            int jt0 = j0 + n16, jt1 = j0 + 16 + n16;
            float f0 = s0[e] * scale +
                       ((rj0 == ri[e] && jt0 != itok[e]) ? ba[e] : 0.f);
            float f1 = s1[e] * scale +
                       ((rj1 == ri[e] && jt1 != itok[e]) ? ba[e] : 0.f);
            if (mk0 == 0) f0 = -1e9f;
            if (mk1 == 0) f1 = -1e9f;
            float rm = redmax16(fmaxf(f0, f1));
            float rn = fmaxf(mr[e], rm);
            float cf = __expf(mr[e] - rn);
            float p0 = __expf(f0 - rn), p1 = __expf(f1 - rn);
            float rs = redsum16(p0 + p1);
            lr[e] = lr[e] * cf + rs;
            mr[e] = rn;
            acc[0][e] *= cf; acc[1][e] *= cf; acc[2][e] *= cf; acc[3][e] *= cf;
            int prow = e + 8 * hf;
            Pst[wave * 512 + prow * 32 + n16]      = (__bf16)p0;
            Pst[wave * 512 + prow * 32 + 16 + n16] = (__bf16)p1;
        }
        // intra-wave LDS RAW: DS ops are in-order; wait for data visibility
        asm volatile("s_wait_dscnt 0" ::: "memory");

        // P back as A-layout 16x32
        const __bf16* pp = &Pst[wave * 512 + n16 * 32];
    v16bf pa = cat8(ld8(pp + 16 * hf), ld8(pp + 16 * hf + 8));

        // PV: 4 output d-tiles, B from transposed V (contiguous j per lane)
#pragma unroll
        for (int t = 0; t < 4; ++t) {
            const __bf16* vb = &Vt[buf][(t * 16 + n16) * 32 + 16 * hf];
            v16bf bv = cat8(ld8(vb), ld8(vb + 8));
            acc[t] = wmma_bf16(pa, bv, acc[t]);
        }
    }

    // normalize and store O as bf16 [B,T,D]
#pragma unroll
    for (int t = 0; t < 4; ++t) {
#pragma unroll
        for (int e = 0; e < 8; ++e) {
            float val = acc[t][e] / lr[e];
            O[((size_t)bb * Tc + itok[e]) * Dc + hh * HDc + t * 16 + n16] =
                (__bf16)val;
        }
    }
}

// ---------------------------------------------------------------------------
// host-side launch
// ---------------------------------------------------------------------------
extern "C" void kernel_launch(void* const* d_in, const int* in_sizes, int n_in,
                              void* d_out, int out_size, void* d_ws, size_t ws_size,
                              hipStream_t stream) {
    (void)in_sizes; (void)n_in; (void)out_size; (void)ws_size;
    const float* x     = (const float*)d_in[0];
    const int*   amask = (const int*)d_in[1];
    const int*   rel   = (const int*)d_in[2];
    const float* q_w   = (const float*)d_in[3];
    const float* q_b   = (const float*)d_in[4];
    const float* k_w   = (const float*)d_in[5];
    const float* k_b   = (const float*)d_in[6];
    const float* v_w   = (const float*)d_in[7];
    const float* v_b   = (const float*)d_in[8];
    const float* o_w   = (const float*)d_in[9];
    const float* o_b   = (const float*)d_in[10];
    const float* hop0  = (const float*)d_in[11];
    const float* hop1  = (const float*)d_in[12];
    const float* pw    = (const float*)d_in[14];
    float* out = (float*)d_out;

    // workspace carve-up
    char*  base  = (char*)d_ws;
    float* wsm   = (float*)base;                 // softmaxed path weights
    float* biasA = (float*)(base + 256);         // B*T f32
    const size_t SZ_X = (size_t)BTc * Dc;        // elements
    const size_t SZ_W = (size_t)Dc * Dc;
    __bf16* xbf = (__bf16*)(base + 65536);
    __bf16* qwb = xbf + SZ_X;
    __bf16* kwb = qwb + SZ_W;
    __bf16* vwb = kwb + SZ_W;
    __bf16* owb = vwb + SZ_W;
    __bf16* Qb  = owb + SZ_W;
    __bf16* Kb  = Qb + SZ_X;
    __bf16* Vb  = Kb + SZ_X;
    __bf16* Ob  = Vb + SZ_X;

    // 1) softmax(path_weights)
    mhra_softmax3_kernel<<<1, 32, 0, stream>>>(pw, wsm);
    // 2) per-batch histogram + collapsed row bias
    mhra_bias_prep_kernel<<<Bc, 256, 0, stream>>>(rel, hop0, hop1, wsm, biasA);
    // 3) bf16 conversions
    {
        int n = (int)SZ_X;
        mhra_cvt_bf16_kernel<<<(n + 255) / 256, 256, 0, stream>>>(x, xbf, n);
        int m = (int)SZ_W;
        mhra_cvt_bf16_kernel<<<(m + 255) / 256, 256, 0, stream>>>(q_w, qwb, m);
        mhra_cvt_bf16_kernel<<<(m + 255) / 256, 256, 0, stream>>>(k_w, kwb, m);
        mhra_cvt_bf16_kernel<<<(m + 255) / 256, 256, 0, stream>>>(v_w, vwb, m);
        mhra_cvt_bf16_kernel<<<(m + 255) / 256, 256, 0, stream>>>(o_w, owb, m);
    }
    // 4) Q/K/V projections (WMMA), written as bf16 [B,H,T,HD]
    dim3 ggrid(BTc / 128, Dc / 64);
    mhra_gemm_kernel<<<ggrid, 128, 0, stream>>>(xbf, qwb, q_b, Qb, nullptr, 0);
    mhra_gemm_kernel<<<ggrid, 128, 0, stream>>>(xbf, kwb, k_b, Kb, nullptr, 0);
    mhra_gemm_kernel<<<ggrid, 128, 0, stream>>>(xbf, vwb, v_b, Vb, nullptr, 0);
    // 5) flash attention with relation bias -> Ob bf16 [B,T,D]
    mhra_flash_attn_kernel<<<dim3(Tc / 64, Bc * Hc), 128, 0, stream>>>(
        Qb, Kb, Vb, rel, amask, biasA, Ob);
    // 6) output projection (WMMA) -> f32 d_out [B,T,D]
    mhra_gemm_kernel<<<ggrid, 128, 0, stream>>>(Ob, owb, o_b, nullptr, out, 1);
}
